// SelfAttn1d_46394236731953
// MI455X (gfx1250) — compile-verified, matrix operand order checked by reading, not run on hardware
//
#include <hip/hip_runtime.h>
#include <math.h>

// ---------------------------------------------------------------- types
typedef unsigned short u16;
typedef __attribute__((ext_vector_type(16))) __bf16 v16bf;
typedef __attribute__((ext_vector_type(8)))  float  v8f;
typedef __attribute__((ext_vector_type(4)))  int    v4i_;

#define B_   2
#define S_   2048
#define E_   1024
#define H_   16
#define DH_  64
#define NROW (B_ * S_)          // 4096 rows

// ---------------- CDNA5 feature probes (guarded so compile never regresses)
#if __has_builtin(__builtin_amdgcn_global_load_async_to_lds_b128)
#define HAVE_ASYNC 1
#else
#define HAVE_ASYNC 0
#endif
#if __has_builtin(__builtin_amdgcn_permlane16)
#define HAVE_PERMLANE 1
#else
#define HAVE_PERMLANE 0
#endif

// async global->LDS 16B copy (ASYNCcnt-tracked) with sync fallback
__device__ __forceinline__ void async_cp16(const u16* gsrc, u16* ldst) {
#if HAVE_ASYNC
    __builtin_amdgcn_global_load_async_to_lds_b128(
        (v4i_*)gsrc, (v4i_*)ldst, 0, 0);
#else
    *(uint4*)ldst = *(const uint4*)gsrc;
#endif
}

__device__ __forceinline__ void async_wait() {
#if HAVE_ASYNC
#if __has_builtin(__builtin_amdgcn_s_wait_asynccnt)
    __builtin_amdgcn_s_wait_asynccnt(0);
#else
    asm volatile("s_wait_asynccnt 0" ::: "memory");
#endif
#endif
}

// xor-lane-shuffle within 16-lane rows: v_permlane16_b32 (pure VALU, no LDS)
constexpr unsigned selword(int base, int mask) {
    unsigned v = 0;
    for (int i = 0; i < 8; ++i) v |= ((unsigned)((base + i) ^ mask) & 0xFu) << (4 * i);
    return v;
}
template <int MASK>
__device__ __forceinline__ float xshuf(float x) {
#if HAVE_PERMLANE
    return __uint_as_float(__builtin_amdgcn_permlane16(
        __float_as_uint(x), __float_as_uint(x),
        selword(0, MASK), selword(8, MASK), false, false));
#else
    return __shfl_xor(x, MASK, 32);
#endif
}
__device__ __forceinline__ float rowmax16(float x) {
    x = fmaxf(x, xshuf<1>(x)); x = fmaxf(x, xshuf<2>(x));
    x = fmaxf(x, xshuf<4>(x)); x = fmaxf(x, xshuf<8>(x));
    return x;
}
__device__ __forceinline__ float rowsum16(float x) {
    x += xshuf<1>(x); x += xshuf<2>(x); x += xshuf<4>(x); x += xshuf<8>(x);
    return x;
}

// fp32 -> bf16 (round-to-nearest-even, NaN edge ignored)
__device__ __forceinline__ u16 f2bf(float f) {
    union { float f; unsigned u; } v; v.f = f;
    unsigned r = v.u + 0x7FFFu + ((v.u >> 16) & 1u);
    return (u16)(r >> 16);
}

// Load a 16x32 A-fragment (or 32x16 B-fragment) row for one lane.
// ISA 16-bit layout: lane L holds row m=L&15; element h maps to
// k = (h>>3)*16 + (L>>4)*8 + (h&7)  -> two contiguous 8-element runs.
__device__ __forceinline__ v16bf ld_frag(const u16* row, int hi) {
    union { uint4 q; __bf16 b[8]; } lo, hh;
    lo.q = *(const uint4*)(row + hi * 8);
    hh.q = *(const uint4*)(row + 16 + hi * 8);
    v16bf f;
#pragma unroll
    for (int i = 0; i < 8; ++i) { f[i] = lo.b[i]; f[8 + i] = hh.b[i]; }
    return f;
}

// ------------------------------------------------- 1) LN + GELU + encodings
__global__ void k_pre(const float* __restrict__ x, const float* __restrict__ mul,
                      const float* __restrict__ add, const float* __restrict__ lw,
                      const float* __restrict__ lb,
                      u16* __restrict__ qkin, u16* __restrict__ val) {
    const int row = blockIdx.x;
    const int t   = threadIdx.x;                 // 256 threads, 4 elems each
    const float4 xv = ((const float4*)(x + (size_t)row * E_))[t];
    float s  = xv.x + xv.y + xv.z + xv.w;
    float s2 = xv.x * xv.x + xv.y * xv.y + xv.z * xv.z + xv.w * xv.w;
#pragma unroll
    for (int o = 16; o > 0; o >>= 1) {
        s  += __shfl_xor(s,  o, 32);
        s2 += __shfl_xor(s2, o, 32);
    }
    __shared__ float rs[8], rs2[8];
    const int w = t >> 5, ln = t & 31;
    if (ln == 0) { rs[w] = s; rs2[w] = s2; }
    __syncthreads();
    s = 0.f; s2 = 0.f;
#pragma unroll
    for (int i = 0; i < 8; ++i) { s += rs[i]; s2 += rs2[i]; }
    const float mean = s * (1.f / E_);
    const float var  = s2 * (1.f / E_) - mean * mean;
    const float rstd = rsqrtf(var + 1e-5f);

    const float4 mv  = ((const float4*)(mul + (size_t)row * E_))[t];
    const float4 av  = ((const float4*)(add + (size_t)row * E_))[t];
    const float4 lwv = ((const float4*)lw)[t];
    const float4 lbv = ((const float4*)lb)[t];
    float xs[4] = { xv.x,  xv.y,  xv.z,  xv.w  };
    float ms[4] = { mv.x,  mv.y,  mv.z,  mv.w  };
    float as[4] = { av.x,  av.y,  av.z,  av.w  };
    float ws[4] = { lwv.x, lwv.y, lwv.z, lwv.w };
    float bs[4] = { lbv.x, lbv.y, lbv.z, lbv.w };
    union { u16 s[4]; uint2 v; } oq, ov;
#pragma unroll
    for (int j = 0; j < 4; ++j) {
        float d = (xs[j] - mean) * rstd * ws[j] + bs[j];
        float g = 0.5f * d * (1.f + erff(d * 0.70710678118654752f)); // exact GELU
        ov.s[j] = f2bf(g);
        oq.s[j] = f2bf(g * ms[j] + as[j]);
    }
    *(uint2*)(val  + (size_t)row * E_ + t * 4) = ov.v;
    *(uint2*)(qkin + (size_t)row * E_ + t * 4) = oq.v;
}

// ------------------------------------------------- 2) fp32 -> bf16 weights
__global__ void k_cvt(const float* __restrict__ s, u16* __restrict__ d) {
    const size_t i = ((size_t)blockIdx.x * 256 + threadIdx.x) * 4;
    const float4 v = *(const float4*)(s + i);
    union { u16 s[4]; uint2 p; } o;
    o.s[0] = f2bf(v.x); o.s[1] = f2bf(v.y); o.s[2] = f2bf(v.z); o.s[3] = f2bf(v.w);
    *(uint2*)(d + i) = o.p;
}

// ------------------------------------------------- 3) WMMA GEMM  C = A * W^T
// A: [M,K] bf16 row-major, W: [Nout,K] bf16 row-major (NT GEMM).
// MODE 0: Cb[row,col] = bf16((acc + bias[col]) * alpha)
// MODE 1: Cf[row,col] = acc + bias[col] + resid[row,col]   (fp32 out)
template <int MODE>
__global__ void k_gemm(const u16* __restrict__ A, const u16* __restrict__ W,
                       const float* __restrict__ bias, float alpha,
                       u16* __restrict__ Cb, float* __restrict__ Cf,
                       const float* __restrict__ resid, int M, int Nout, int K) {
    __shared__ __align__(16) u16 As[64][40];
    __shared__ __align__(16) u16 Bs[128][40];
    const int m0 = blockIdx.y * 64, n0 = blockIdx.x * 128;
    const int t = threadIdx.x;
    const int wave = t >> 5, lane = t & 31, ln = lane & 15, hi = lane >> 4;
    const int wm = (wave & 1) * 32, wn = (wave >> 1) * 32;

    v8f acc[2][2] = {};
    for (int k0 = 0; k0 < K; k0 += 32) {
        // stage A tile 64x32 + B tile 128x32 via ASYNCcnt-tracked LDS DMA
        const int ra = t >> 2, ca = (t & 3) * 8;
        const u16* srcA = A + (size_t)(m0 + ra) * K + k0 + ca;
        async_cp16(srcA, &As[ra][ca]);
#pragma unroll
        for (int i = 0; i < 2; ++i) {
            const int cid = t + i * 256;
            const int r = cid >> 2, c = (cid & 3) * 8;
            async_cp16(W + (size_t)(n0 + r) * K + k0 + c, &Bs[r][c]);
        }
        if (k0 + 32 < K) __builtin_prefetch(srcA + 32, 0, 1);   // global_prefetch
        async_wait();
        __syncthreads();

        v16bf a[2], b[2];
#pragma unroll
        for (int mi = 0; mi < 2; ++mi) a[mi] = ld_frag(&As[wm + mi * 16 + ln][0], hi);
#pragma unroll
        for (int ni = 0; ni < 2; ++ni) b[ni] = ld_frag(&Bs[wn + ni * 16 + ln][0], hi);
#pragma unroll
        for (int mi = 0; mi < 2; ++mi)
#pragma unroll
            for (int ni = 0; ni < 2; ++ni)
                acc[mi][ni] = __builtin_amdgcn_wmma_f32_16x16x32_bf16(
                    false, a[mi], false, b[ni], (short)0, acc[mi][ni], false, false);
        __syncthreads();
    }
#pragma unroll
    for (int ni = 0; ni < 2; ++ni) {
        const int col = n0 + wn + ni * 16 + ln;
        const float bc = bias[col];
#pragma unroll
        for (int mi = 0; mi < 2; ++mi)
#pragma unroll
            for (int r = 0; r < 8; ++r) {
                const int row = m0 + wm + mi * 16 + r + 8 * hi;
                const size_t idx = (size_t)row * Nout + col;
                const float v = acc[mi][ni][r];
                if (MODE == 0) Cb[idx] = f2bf((v + bc) * alpha);
                else           Cf[idx] = v + bc + resid[idx];
            }
    }
}

// ------------------------------------------------- 4) flash attention
// 128 query rows per block (16 per wave), 32-key tiles, online softmax.
__global__ void k_attn(const u16* __restrict__ Q, const u16* __restrict__ Kb,
                       const u16* __restrict__ Vb, u16* __restrict__ ctx) {
    __shared__ __align__(16) u16 Ks[32][72];       // [key][d]
    __shared__ __align__(16) u16 Vt[64][40];       // [d][key]  (transposed)
    __shared__ __align__(16) u16 Ps[8][16][40];    // per-wave P tile [q][key]

    const int QT = S_ / 128;                       // 16 q-tiles
    const int bid = blockIdx.x;
    const int qt = bid % QT;
    const int hh = (bid / QT) % H_;
    const int bb = bid / (QT * H_);
    const int t = threadIdx.x, wave = t >> 5, lane = t & 31, ln = lane & 15, hi = lane >> 4;
    const int q0 = qt * 128 + wave * 16;
    const size_t rowbase = (size_t)bb * S_;
    const int cofs = hh * DH_;

    // Q fragments (1/sqrt(DH) folded in at projection time)
    const u16* qp = Q + (rowbase + q0 + ln) * E_ + cofs;
    v16bf qa[2];
#pragma unroll
    for (int f = 0; f < 2; ++f) qa[f] = ld_frag(qp + f * 32, hi);

    float mrow[8], lrow[8];
#pragma unroll
    for (int r = 0; r < 8; ++r) { mrow[r] = -3.0e38f; lrow[r] = 0.f; }
    v8f o[4] = {};

    for (int key0 = 0; key0 < S_; key0 += 32) {
        __syncthreads();                           // protect tiles still in use
        {   // K tile: async global->LDS DMA; V tile: register transpose
            const int kr = t >> 3, cc = (t & 7) * 8;
            const u16* srcK = Kb + (rowbase + key0 + kr) * E_ + cofs + cc;
            async_cp16(srcK, &Ks[kr][cc]);
            if (key0 + 32 < S_) __builtin_prefetch(srcK + 32 * E_, 0, 1);
            union { uint4 q; u16 s[8]; } u;
            u.q = *(const uint4*)(Vb + (rowbase + key0 + kr) * E_ + cofs + cc);
#pragma unroll
            for (int j = 0; j < 8; ++j) Vt[cc + j][kr] = u.s[j];
        }
        async_wait();
        __syncthreads();

        // S = Q * K^T  (16 q x 32 keys, two 16-col groups, K-dim 64 = 2 wmmas each)
        v8f c0 = {}, c1 = {};
#pragma unroll
        for (int f = 0; f < 2; ++f) {
            c0 = __builtin_amdgcn_wmma_f32_16x16x32_bf16(
                false, qa[f], false, ld_frag(&Ks[ln][0] + f * 32, hi), (short)0, c0, false, false);
            c1 = __builtin_amdgcn_wmma_f32_16x16x32_bf16(
                false, qa[f], false, ld_frag(&Ks[16 + ln][0] + f * 32, hi), (short)0, c1, false, false);
        }

        // online softmax; lanes sharing hi hold the 16 columns of each row
#pragma unroll
        for (int r = 0; r < 8; ++r) {
            const float mx   = rowmax16(fmaxf(c0[r], c1[r]));
            const float mnew = fmaxf(mrow[r], mx);
            const float corr = __expf(mrow[r] - mnew);
            const float p0 = __expf(c0[r] - mnew);
            const float p1 = __expf(c1[r] - mnew);
            const float rsum = rowsum16(p0 + p1);
            lrow[r] = lrow[r] * corr + rsum;
            mrow[r] = mnew;
#pragma unroll
            for (int tt = 0; tt < 4; ++tt) o[tt][r] *= corr;
            Ps[wave][r + 8 * hi][ln]      = f2bf(p0);   // C-layout -> A-layout via LDS
            Ps[wave][r + 8 * hi][16 + ln] = f2bf(p1);
        }
        __syncthreads();

        // O += P * V   (16 x 64 output, 4 wmmas, K-dim = 32 keys)
        const v16bf pa = ld_frag(&Ps[wave][ln][0], hi);
#pragma unroll
        for (int tt = 0; tt < 4; ++tt) {
            const v16bf vb = ld_frag(&Vt[tt * 16 + ln][0], hi);
            o[tt] = __builtin_amdgcn_wmma_f32_16x16x32_bf16(
                false, pa, false, vb, (short)0, o[tt], false, false);
        }
    }

    float rinv[8];
#pragma unroll
    for (int r = 0; r < 8; ++r) rinv[r] = 1.0f / lrow[r];
#pragma unroll
    for (int tt = 0; tt < 4; ++tt)
#pragma unroll
        for (int r = 0; r < 8; ++r)
            ctx[(rowbase + q0 + r + 8 * hi) * E_ + cofs + tt * 16 + ln] =
                f2bf(o[tt][r] * rinv[r]);
}

// ---------------------------------------------------------------- launcher
extern "C" void kernel_launch(void* const* d_in, const int* in_sizes, int n_in,
                              void* d_out, int out_size, void* d_ws, size_t ws_size,
                              hipStream_t stream) {
    (void)in_sizes; (void)n_in; (void)out_size; (void)ws_size;
    const float* in_feats = (const float*)d_in[0];
    const float* mul_encs = (const float*)d_in[1];
    const float* add_encs = (const float*)d_in[2];
    const float* ln_w     = (const float*)d_in[3];
    const float* ln_b     = (const float*)d_in[4];
    const float* w_qkv    = (const float*)d_in[5];   // [3E, E]
    const float* b_qkv    = (const float*)d_in[6];   // [3E]
    const float* w_out    = (const float*)d_in[7];   // [E, E]
    const float* b_out    = (const float*)d_in[8];   // [E]
    float* out = (float*)d_out;

    char* ws = (char*)d_ws;
    const size_t SZ = (size_t)NROW * E_ * sizeof(u16);   // 8 MB per activation buf
    u16* qkin = (u16*)(ws);
    u16* val  = (u16*)(ws + 1 * SZ);
    u16* Qb   = (u16*)(ws + 2 * SZ);
    u16* Kbuf = (u16*)(ws + 3 * SZ);
    u16* Vbuf = (u16*)(ws + 4 * SZ);
    u16* wqkv = (u16*)(ws + 5 * SZ);                     // 3E*E bf16 (6 MB)
    u16* wo   = (u16*)(ws + 5 * SZ + (size_t)3 * E_ * E_ * sizeof(u16));
    u16* ctx  = val;   // values consumed by the V GEMM before attention runs

    // weights fp32 -> bf16
    k_cvt<<<(3 * E_ * E_) / 1024, 256, 0, stream>>>(w_qkv, wqkv);
    k_cvt<<<(E_ * E_) / 1024,     256, 0, stream>>>(w_out, wo);

    // LN + GELU + encoding modulation
    k_pre<<<NROW, 256, 0, stream>>>(in_feats, mul_encs, add_encs, ln_w, ln_b, qkin, val);

    // QKV projections (alpha folds 1/sqrt(DH) into Q)
    const dim3 gg(E_ / 128, NROW / 64);
    const float qscale = 1.0f / sqrtf((float)DH_);
    k_gemm<0><<<gg, 256, 0, stream>>>(qkin, wqkv,                       b_qkv,          qscale, Qb,   nullptr, nullptr, NROW, E_, E_);
    k_gemm<0><<<gg, 256, 0, stream>>>(qkin, wqkv + (size_t)E_ * E_,     b_qkv + E_,     1.0f,   Kbuf, nullptr, nullptr, NROW, E_, E_);
    k_gemm<0><<<gg, 256, 0, stream>>>(val,  wqkv + (size_t)2 * E_ * E_, b_qkv + 2 * E_, 1.0f,   Vbuf, nullptr, nullptr, NROW, E_, E_);

    // attention: B*H*(S/128) blocks
    k_attn<<<B_ * H_ * (S_ / 128), 256, 0, stream>>>(Qb, Kbuf, Vbuf, ctx);

    // output projection + bias + residual (fp32 out)
    k_gemm<1><<<gg, 256, 0, stream>>>(ctx, wo, b_out, 1.0f, nullptr, out, in_feats, NROW, E_, E_);
}
